// CrossAttention_86981677679312
// MI455X (gfx1250) — matrix operand order repaired and checked by reading, not compiled, not run on hardware
//
#include <hip/hip_runtime.h>
#include <hip/hip_bf16.h>
#include <stdint.h>

#define CDIM 512
#define HD   64
#define NH   8
#define BATCH 2
#define SEQ  4096
#define ROWS (BATCH*SEQ)   /* 8192 */
#define KVCHUNK 64
#define NIT (SEQ / KVCHUNK)

typedef __attribute__((ext_vector_type(16))) __bf16 v16bf;
typedef __attribute__((ext_vector_type(8)))  float  v8f;
typedef unsigned int u32x4 __attribute__((ext_vector_type(4)));
typedef int i32x4 __attribute__((ext_vector_type(4)));
typedef int i32x8 __attribute__((ext_vector_type(8)));

union FragU { v16bf v; uint4 u[2]; };

// TDM availability must be evaluated in the DEVICE pass only: amdgcn builtins
// do not exist in the host (x86) pass of HIP's dual compilation.
#if defined(__HIP_DEVICE_COMPILE__)
#if __has_builtin(__builtin_amdgcn_tensor_load_to_lds) && \
    __has_builtin(__builtin_amdgcn_s_wait_tensorcnt)
#define USE_TDM 1
#else
// Decisive probe (device pass only): if this fires, the gfx1250 compiler lacks
// the TDM builtins and we revert to the cooperative-copy fallback next round.
#error "CDNA5 probe: TDM builtins NOT available in DEVICE pass"
#endif
#endif
#ifndef USE_TDM
#define USE_TDM 0   /* host pass: parse-only fallback */
#endif

__device__ __forceinline__ v8f wmma_bf16(v16bf a, v16bf b, v8f c) {
  return __builtin_amdgcn_wmma_f32_16x16x32_bf16(false, a, false, b, (short)0, c,
                                                 false, false);
}

// A operand: 16(M) x 32(K); lanes 0-15 row=lane K=0..7,16..23 ; lanes 16-31 K=8..15,24..31
__device__ __forceinline__ v16bf load_a_frag(const __bf16* base, int ld, int row0,
                                             int k0, int lane) {
  const int r = row0 + (lane & 15);
  const int hf = lane >> 4;
  const __bf16* p = base + (size_t)r * ld + (k0 + hf * 8);
  FragU f;
  f.u[0] = *(const uint4*)(p);
  f.u[1] = *(const uint4*)(p + 16);
  return f.v;
}

// B operand: 32(K) x 16(N), data given transposed (rows = N cols).
// lanes 0-15 col=lane K=0..7,16..23 ; lanes 16-31 col=lane-16 K=8..15,24..31
__device__ __forceinline__ v16bf load_b_frag(const __bf16* baseT, int ldT, int col0,
                                             int k0, int lane) {
  const int n = lane & 15;
  const int hf = lane >> 4;
  const __bf16* p = baseT + (size_t)(col0 + n) * ldT + (k0 + hf * 8);
  FragU f;
  f.u[0] = *(const uint4*)(p);
  f.u[1] = *(const uint4*)(p + 16);
  return f.v;
}

// ------------------------------------------------- TDM 2D tile load (D# per ISA §8)
#if USE_TDM
__device__ __forceinline__ void tdm_load_2d(void* ldsDst, const void* gsrc,
                                            unsigned t_d0, unsigned t_d1,
                                            unsigned tile_d0, unsigned tile_d1,
                                            unsigned long long stride0) {
  const unsigned lds_addr = (unsigned)(size_t)ldsDst;           // addr[31:0] = LDS offset
  const unsigned long long ga = (unsigned long long)(size_t)gsrc;
  const u32x4 g0 = {                       // group0: 128b
      1u,                                  // count=1, user descriptor
      lds_addr,                            // bits 63:32 lds_addr
      (unsigned)(ga & 0xffffffffull),      // global_addr[31:0]
      (unsigned)((ga >> 32) & 0x01ffffffull) | 0x80000000u };  // ga[56:32] | type=2
  const i32x8 g1 = {                       // group1: 256b
      (int)(1u << 16),                                        // data_size=1 (2B)
      (int)((t_d0 & 0xffffu) << 16),                          // tensor_dim0[15:0]
      (int)((t_d0 >> 16) | ((t_d1 & 0xffffu) << 16)),         // dim0[31:16]|dim1[15:0]
      (int)((t_d1 >> 16) | ((tile_d0 & 0xffffu) << 16)),      // dim1[31:16]|tile_dim0
      (int)(tile_d1 & 0xffffu),                               // tile_dim1, tile_dim2=0
      (int)(stride0 & 0xffffffffull),                         // dim0_stride[31:0]
      (int)((stride0 >> 32) & 0xffffull),                     // dim0_stride[47:32]
      0 };
  const i32x4 gz = {0, 0, 0, 0};
#if __clang_major__ >= 23
  const i32x8 gz8 = {0, 0, 0, 0, 0, 0, 0, 0};
  __builtin_amdgcn_tensor_load_to_lds(g0, g1, gz, gz, gz8, 0);
#else
  __builtin_amdgcn_tensor_load_to_lds(g0, g1, gz, gz, 0);
#endif
}
#endif

// ---------------------------------------------------------------- prep kernels
__global__ void cvt_bf16_kernel(const float* __restrict__ in, __bf16* __restrict__ out,
                                int n) {
  int i = blockIdx.x * blockDim.x + threadIdx.x;
  if (i < n) out[i] = (__bf16)in[i];
}

__global__ void transpose_cvt_kernel(const float* __restrict__ W,
                                     __bf16* __restrict__ WT) {
  __shared__ float tile[16][17];
  int r = blockIdx.y * 16 + threadIdx.y;
  int c = blockIdx.x * 16 + threadIdx.x;
  tile[threadIdx.y][threadIdx.x] = W[r * CDIM + c];
  __syncthreads();
  int rr = blockIdx.x * 16 + threadIdx.y;
  int cc = blockIdx.y * 16 + threadIdx.x;
  WT[rr * CDIM + cc] = (__bf16)tile[threadIdx.x][threadIdx.y];
}

__global__ void pack_mask_kernel(const unsigned char* __restrict__ m,
                                 unsigned* __restrict__ bits, int nwords) {
  int w = blockIdx.x * blockDim.x + threadIdx.x;
  if (w < nwords) {
    const unsigned char* p = m + (size_t)w * 32;
    unsigned v = 0;
#pragma unroll
    for (int j = 0; j < 32; ++j) v |= (p[j] ? 1u : 0u) << j;
    bits[w] = v;
  }
}

// -------------------------------------------------------- projection GEMM (+RoPE)
// mode 0: per-head (b,h,seq,64) bf16 with 2D RoPE (Q/K)
// mode 1: per-head transposed (b,h,64,seq) bf16   (V)
// mode 2: row-major (rows,512) f32                (O-projection)
__global__ __launch_bounds__(256)
void proj_gemm_kernel(const __bf16* __restrict__ A, const __bf16* __restrict__ BT,
                      const float* __restrict__ bias, const int* __restrict__ pos,
                      __bf16* __restrict__ outHead, __bf16* __restrict__ outVT,
                      float* __restrict__ outF, int mode, int seqLen) {
  const int lane = threadIdx.x & 31;
  const int wave = threadIdx.x >> 5;      // wave == head (8 waves x 64 cols)
  const int rowBase = blockIdx.x * 64;
  const int col0 = wave * 64;
  const int nloc = lane & 15;
  const int hf = lane >> 4;

  v8f acc[4][4];
#pragma unroll
  for (int rt = 0; rt < 4; ++rt)
#pragma unroll
    for (int ct = 0; ct < 4; ++ct)
#pragma unroll
      for (int i = 0; i < 8; ++i) acc[rt][ct][i] = 0.0f;

  for (int k0 = 0; k0 < CDIM; k0 += 32) {
    v16bf af[4], bfm[4];
#pragma unroll
    for (int rt = 0; rt < 4; ++rt)
      af[rt] = load_a_frag(A, CDIM, rowBase + rt * 16, k0, lane);
#pragma unroll
    for (int ct = 0; ct < 4; ++ct)
      bfm[ct] = load_b_frag(BT, CDIM, col0 + ct * 16, k0, lane);
#pragma unroll
    for (int rt = 0; rt < 4; ++rt)
#pragma unroll
      for (int ct = 0; ct < 4; ++ct)
        acc[rt][ct] = wmma_bf16(af[rt], bfm[ct], acc[rt][ct]);
  }

#pragma unroll
  for (int ct = 0; ct < 4; ++ct) {
    const float bv = bias[col0 + ct * 16 + nloc];
#pragma unroll
    for (int rt = 0; rt < 4; ++rt)
#pragma unroll
      for (int i = 0; i < 8; ++i) acc[rt][ct][i] += bv;
  }

  if (mode == 0) {
    const float invf = __powf(100.0f, -(float)nloc * (1.0f / 16.0f));
#pragma unroll
    for (int rt = 0; rt < 4; ++rt) {
#pragma unroll
      for (int i = 0; i < 8; ++i) {
        const int row = rowBase + rt * 16 + i + hf * 8;
        const float py = (float)pos[row * 2 + 0];
        const float px = (float)pos[row * 2 + 1];
        {
          const float ang = py * invf;
          const float cs = __cosf(ang), sn = __sinf(ang);
          const float a0 = acc[rt][0][i], a1 = acc[rt][1][i];
          acc[rt][0][i] = a0 * cs - a1 * sn;
          acc[rt][1][i] = a1 * cs + a0 * sn;
        }
        {
          const float ang = px * invf;
          const float cs = __cosf(ang), sn = __sinf(ang);
          const float a0 = acc[rt][2][i], a1 = acc[rt][3][i];
          acc[rt][2][i] = a0 * cs - a1 * sn;
          acc[rt][3][i] = a1 * cs + a0 * sn;
        }
        const int b = row / seqLen;
        const int n = row - b * seqLen;
        const size_t base = (((size_t)b * NH + wave) * seqLen + n) * HD;
#pragma unroll
        for (int ct = 0; ct < 4; ++ct)
          outHead[base + ct * 16 + nloc] = (__bf16)acc[rt][ct][i];
      }
    }
  } else if (mode == 1) {
#pragma unroll
    for (int rt = 0; rt < 4; ++rt)
#pragma unroll
      for (int i = 0; i < 8; ++i) {
        const int row = rowBase + rt * 16 + i + hf * 8;
        const int b = row / seqLen;
        const int n = row - b * seqLen;
#pragma unroll
        for (int ct = 0; ct < 4; ++ct) {
          const int d = ct * 16 + nloc;
          outVT[(((size_t)b * NH + wave) * HD + d) * seqLen + n] =
              (__bf16)acc[rt][ct][i];
        }
      }
  } else {
#pragma unroll
    for (int rt = 0; rt < 4; ++rt)
#pragma unroll
      for (int i = 0; i < 8; ++i) {
        const int row = rowBase + rt * 16 + i + hf * 8;
#pragma unroll
        for (int ct = 0; ct < 4; ++ct)
          outF[(size_t)row * CDIM + col0 + ct * 16 + nloc] = acc[rt][ct][i];
      }
  }
}

// ------------------------------------------------------------- flash attention
__global__ __launch_bounds__(256)
void flash_attn_kernel(const __bf16* __restrict__ Q, const __bf16* __restrict__ K,
                       const __bf16* __restrict__ VT,
                       const unsigned* __restrict__ mbits,
                       __bf16* __restrict__ outA) {
  __shared__ __align__(16) __bf16 Kb[2][KVCHUNK * HD];   // 2 x 8KB
  __shared__ __align__(16) __bf16 Vb[2][HD * KVCHUNK];   // 2 x 8KB
  __shared__ __align__(16) __bf16 Pb[8][16 * KVCHUNK];   // 16KB (per-wave P)

  const int lane = threadIdx.x & 31;
  const int wave = threadIdx.x >> 5;
  const int bh = blockIdx.y;
  const int b = bh >> 3, h = bh & 7;
  const int q0 = blockIdx.x * 128 + wave * 16;
  const int nloc = lane & 15, hf = lane >> 4;

  const __bf16* Qp = Q + (size_t)bh * SEQ * HD;
  const __bf16* Kp = K + (size_t)bh * SEQ * HD;
  const __bf16* Vp = VT + (size_t)bh * HD * SEQ;
  const unsigned* mb = mbits + (size_t)b * SEQ * (SEQ / 32);

  // Q fragments register-resident for the whole kv loop
  const v16bf qf0 = load_a_frag(Qp, HD, q0, 0, lane);
  const v16bf qf1 = load_a_frag(Qp, HD, q0, 32, lane);

  v8f o[4];
#pragma unroll
  for (int t = 0; t < 4; ++t)
#pragma unroll
    for (int i = 0; i < 8; ++i) o[t][i] = 0.0f;
  float mrow[8], lrow[8];
#pragma unroll
  for (int i = 0; i < 8; ++i) { mrow[i] = -3.0e38f; lrow[i] = 0.0f; }

  __bf16* plds = Pb[wave];
  const float scale = 0.125f;  // 1/sqrt(64)

#if USE_TDM
  if (wave == 0) {  // stage chunk 0: K rows [0,64), V^T columns [0,64)
    tdm_load_2d(&Kb[0][0], Kp, HD, KVCHUNK, HD, KVCHUNK, HD);
    tdm_load_2d(&Vb[0][0], Vp, KVCHUNK, HD, KVCHUNK, HD, SEQ);
  }
#endif

  for (int it = 0; it < NIT; ++it) {
    const int kv = it * KVCHUNK;
    __syncthreads();  // all waves done reading the buffer about to be overwritten
#if USE_TDM
    const int cur = it & 1;
    if (wave == 0) {
      if (it + 1 < NIT) {  // prefetch next chunk, then wait for current pair only
        const int nkv = kv + KVCHUNK;
        tdm_load_2d(&Kb[(it + 1) & 1][0], Kp + (size_t)nkv * HD, HD, KVCHUNK, HD,
                    KVCHUNK, HD);
        tdm_load_2d(&Vb[(it + 1) & 1][0], Vp + nkv, KVCHUNK, HD, KVCHUNK, HD, SEQ);
        __builtin_amdgcn_s_wait_tensorcnt(2);
      } else {
        __builtin_amdgcn_s_wait_tensorcnt(0);
      }
    }
#else
    const int cur = 0;
    {  // cooperative fallback copy (host parse / no-TDM device)
      const int tid = threadIdx.x;
#pragma unroll
      for (int j = 0; j < 2; ++j) {
        const int idx = tid + j * 256;            // 512 uint4 per 8KB chunk
        ((uint4*)&Kb[0][0])[idx] = *(const uint4*)(Kp + (size_t)kv * HD + idx * 8);
        const int row = idx >> 3, c = idx & 7;    // V^T: 64 rows x 8 uint4
        ((uint4*)&Vb[0][0])[idx] = *(const uint4*)(Vp + (size_t)row * SEQ + kv + c * 8);
      }
    }
#endif
    __syncthreads();  // staged data visible to all waves
    const __bf16* Kl = &Kb[cur][0];
    const __bf16* Vl = &Vb[cur][0];

    // S = Q @ K^T : four 16x16 tiles over kv, K-dim 64 (2 WMMAs each)
    v8f s[4];
#pragma unroll
    for (int t = 0; t < 4; ++t) {
      v8f st;
#pragma unroll
      for (int i = 0; i < 8; ++i) st[i] = 0.0f;
      st = wmma_bf16(qf0, load_b_frag(Kl, HD, t * 16, 0, lane), st);
      st = wmma_bf16(qf1, load_b_frag(Kl, HD, t * 16, 32, lane), st);
      s[t] = st;
    }
    // scale + mask from packed bits (2 dword loads per row per 64 kv)
#pragma unroll
    for (int i = 0; i < 8; ++i) {
      const int qr = q0 + i + hf * 8;
      const unsigned w0 = mb[(size_t)qr * (SEQ / 32) + (kv >> 5)];
      const unsigned w1 = mb[(size_t)qr * (SEQ / 32) + (kv >> 5) + 1];
      s[0][i] = ((w0 >> nloc) & 1u) ? -1.0e30f : s[0][i] * scale;
      s[1][i] = ((w0 >> (nloc + 16)) & 1u) ? -1.0e30f : s[1][i] * scale;
      s[2][i] = ((w1 >> nloc) & 1u) ? -1.0e30f : s[2][i] * scale;
      s[3][i] = ((w1 >> (nloc + 16)) & 1u) ? -1.0e30f : s[3][i] * scale;
    }
    // online softmax (row stats per lane, reduced across the 16-lane half)
    float alpha[8];
#pragma unroll
    for (int i = 0; i < 8; ++i) {
      float mx = fmaxf(fmaxf(s[0][i], s[1][i]), fmaxf(s[2][i], s[3][i]));
      mx = fmaxf(mx, __shfl_xor(mx, 1, 32));
      mx = fmaxf(mx, __shfl_xor(mx, 2, 32));
      mx = fmaxf(mx, __shfl_xor(mx, 4, 32));
      mx = fmaxf(mx, __shfl_xor(mx, 8, 32));
      const float mnew = fmaxf(mrow[i], mx);
      alpha[i] = __expf(mrow[i] - mnew);
      mrow[i] = mnew;
      const float p0 = __expf(s[0][i] - mnew);
      const float p1 = __expf(s[1][i] - mnew);
      const float p2 = __expf(s[2][i] - mnew);
      const float p3 = __expf(s[3][i] - mnew);
      s[0][i] = p0; s[1][i] = p1; s[2][i] = p2; s[3][i] = p3;
      float r = (p0 + p1) + (p2 + p3);
      r += __shfl_xor(r, 1, 32);
      r += __shfl_xor(r, 2, 32);
      r += __shfl_xor(r, 4, 32);
      r += __shfl_xor(r, 8, 32);
      lrow[i] = lrow[i] * alpha[i] + r;
    }
#pragma unroll
    for (int t = 0; t < 4; ++t)
#pragma unroll
      for (int i = 0; i < 8; ++i) o[t][i] *= alpha[i];

    // P: acc layout -> A-operand layout via per-wave LDS (same-wave DS is in-order)
#pragma unroll
    for (int t = 0; t < 4; ++t)
#pragma unroll
      for (int i = 0; i < 8; ++i) {
        const int m = i + hf * 8;
        plds[m * KVCHUNK + t * 16 + nloc] = (__bf16)s[t][i];
      }
    asm volatile("" ::: "memory");
    const v16bf pf0 = load_a_frag(plds, KVCHUNK, 0, 0, lane);
    const v16bf pf1 = load_a_frag(plds, KVCHUNK, 0, 32, lane);

    // O += P(16x64) @ V(64x64) from staged V^T in LDS
#pragma unroll
    for (int t = 0; t < 4; ++t) {
      o[t] = wmma_bf16(pf0, load_b_frag(Vl, KVCHUNK, t * 16, 0, lane), o[t]);
      o[t] = wmma_bf16(pf1, load_b_frag(Vl, KVCHUNK, t * 16, 32, lane), o[t]);
    }
  }

  // epilogue: normalize, emit bf16 in (B,N,C) head-concat layout
#pragma unroll
  for (int i = 0; i < 8; ++i) {
    const float inv = (lrow[i] > 0.0f) ? (1.0f / lrow[i]) : 0.0f;
    const int qr = q0 + i + hf * 8;
    const size_t rowbase = ((size_t)b * SEQ + qr) * CDIM + h * HD;
#pragma unroll
    for (int t = 0; t < 4; ++t)
      outA[rowbase + t * 16 + nloc] = (__bf16)(o[t][i] * inv);
  }
}

// ------------------------------------------------------------------- launcher
extern "C" void kernel_launch(void* const* d_in, const int* in_sizes, int n_in,
                              void* d_out, int out_size, void* d_ws, size_t ws_size,
                              hipStream_t stream) {
  (void)in_sizes; (void)n_in; (void)out_size; (void)ws_size;
  const float* x    = (const float*)d_in[0];
  const float* ctx  = (const float*)d_in[1];
  const int* pos_x  = (const int*)d_in[2];
  const int* pos_c  = (const int*)d_in[3];
  const unsigned char* mask = (const unsigned char*)d_in[4];
  const float* Wq = (const float*)d_in[5];
  const float* bq = (const float*)d_in[6];
  const float* Wk = (const float*)d_in[7];
  const float* bk = (const float*)d_in[8];
  const float* Wv = (const float*)d_in[9];
  const float* bv = (const float*)d_in[10];
  const float* Wo = (const float*)d_in[11];
  const float* bo = (const float*)d_in[12];
  float* out = (float*)d_out;

  char* ws = (char*)d_ws;
  const size_t SZ_ACT = (size_t)ROWS * CDIM * sizeof(__bf16);  // 8 MB
  const size_t SZ_W   = (size_t)CDIM * CDIM * sizeof(__bf16);  // 512 KB
  __bf16* xbf  = (__bf16*)(ws);
  __bf16* cbf  = (__bf16*)(ws + SZ_ACT);
  __bf16* WqT  = (__bf16*)(ws + 2 * SZ_ACT);
  __bf16* WkT  = (__bf16*)(ws + 2 * SZ_ACT + 1 * SZ_W);
  __bf16* WvT  = (__bf16*)(ws + 2 * SZ_ACT + 2 * SZ_W);
  __bf16* WoT  = (__bf16*)(ws + 2 * SZ_ACT + 3 * SZ_W);
  __bf16* Qbf  = (__bf16*)(ws + 2 * SZ_ACT + 4 * SZ_W);
  __bf16* Kbf  = (__bf16*)(ws + 3 * SZ_ACT + 4 * SZ_W);
  __bf16* Vtbf = (__bf16*)(ws + 4 * SZ_ACT + 4 * SZ_W);
  unsigned* mbits = (unsigned*)(ws + 5 * SZ_ACT + 4 * SZ_W);   // 4 MB
  __bf16* attnbf = xbf;  // x dead after Q projection -> reuse

  const int nact = ROWS * CDIM;
  cvt_bf16_kernel<<<(nact + 255) / 256, 256, 0, stream>>>(x, xbf, nact);
  cvt_bf16_kernel<<<(nact + 255) / 256, 256, 0, stream>>>(ctx, cbf, nact);
  dim3 tg(CDIM / 16, CDIM / 16), tb(16, 16);
  transpose_cvt_kernel<<<tg, tb, 0, stream>>>(Wq, WqT);
  transpose_cvt_kernel<<<tg, tb, 0, stream>>>(Wk, WkT);
  transpose_cvt_kernel<<<tg, tb, 0, stream>>>(Wv, WvT);
  transpose_cvt_kernel<<<tg, tb, 0, stream>>>(Wo, WoT);
  const int nwords = ROWS * (SEQ / 32);
  pack_mask_kernel<<<(nwords + 255) / 256, 256, 0, stream>>>(mask, mbits, nwords);

  proj_gemm_kernel<<<ROWS / 64, 256, 0, stream>>>(xbf, WqT, bq, pos_x, Qbf,
                                                  nullptr, nullptr, 0, SEQ);
  proj_gemm_kernel<<<ROWS / 64, 256, 0, stream>>>(cbf, WkT, bk, pos_c, Kbf,
                                                  nullptr, nullptr, 0, SEQ);
  proj_gemm_kernel<<<ROWS / 64, 256, 0, stream>>>(cbf, WvT, bv, nullptr, nullptr,
                                                  Vtbf, nullptr, 1, SEQ);

  dim3 ag(SEQ / 128, BATCH * NH);
  flash_attn_kernel<<<ag, 256, 0, stream>>>(Qbf, Kbf, Vtbf, mbits, attnbf);

  proj_gemm_kernel<<<ROWS / 64, 256, 0, stream>>>(attnbf, WoT, bo, nullptr,
                                                  nullptr, nullptr, out, 2, SEQ);
}
